// SelfAttention_1013612282359
// MI455X (gfx1250) — matrix operand order repaired and checked by reading, not compiled, not run on hardware
//
#include <hip/hip_runtime.h>
#include <hip/hip_bf16.h>

// ---------------------------------------------------------------------------
// Self-attention, MI455X (gfx1250, wave32, WMMA).
// Pipeline: [w->bf16T] -> [QKV gemm, V stored transposed] -> [flash attention]
//           -> [out gemm + bias (f32 out)]
// All intermediates bf16 in d_ws (~72MB), L2-resident (192MB L2).
// GEMMs: one wave = 16x64 output strip (A-frag reused across 4 WMMAs).
// Attention: all three operand streams (Q row, K row, V^T row) contiguous.
// ---------------------------------------------------------------------------

typedef __attribute__((ext_vector_type(16))) __bf16 v16bf;
typedef __attribute__((ext_vector_type(8)))  float  v8f;

static constexpr int Bsz = 4, Lsq = 2048, Cdim = 1024, Dh = 64;
static constexpr int N3C = 3 * Cdim;              // 3072
static constexpr int C2  = 2 * Cdim;              // 2048 (Q|K row width)
static constexpr int Mrows = Bsz * Lsq;           // 8192

__device__ inline v8f wmma_bf16(v16bf a, v16bf b, v8f c) {
  return __builtin_amdgcn_wmma_f32_16x16x32_bf16(
      /*neg_a=*/false, a, /*neg_b=*/false, b,
      /*c_mod=*/(short)0, c, /*reuse_a=*/false, /*reuse_b=*/false);
}

// A-fragment (16x32, MxK). Lane (ln + 16*hf) holds row M=ln; element e holds
// K = k0 + 16*(e>>3) + 8*hf + (e&7). base = row start.
__device__ inline v16bf load_fragA_f32(const float* base, int k0, int hf) {
  v16bf r;
  const float* p = base + k0 + 8 * hf;
#pragma unroll
  for (int i = 0; i < 8; ++i) { r[i] = (__bf16)p[i]; r[i + 8] = (__bf16)p[i + 16]; }
  return r;
}
__device__ inline v16bf load_fragA_bf16(const __bf16* base, int k0, int hf) {
  v16bf r;
  const __bf16* p = base + k0 + 8 * hf;
#pragma unroll
  for (int i = 0; i < 8; ++i) { r[i] = p[i]; r[i + 8] = p[i + 16]; }
  return r;
}
// B-fragment (32x16, KxN). Lane (ln + 16*hf) holds column N=ln; element e holds
// K = k0 + 16*(e>>3) + 8*hf + (e&7). base = column start (N-major store).
__device__ inline v16bf load_fragB_bf16(const __bf16* base, int k0, int hf) {
  v16bf r;
  const __bf16* p = base + k0 + 8 * hf;
#pragma unroll
  for (int i = 0; i < 8; ++i) { r[i] = p[i]; r[i + 8] = p[i + 16]; }
  return r;
}

// --------------------------- weight convert/transpose ----------------------
__global__ void convert_transpose_kernel(const float* __restrict__ w,
                                         __bf16* __restrict__ wT, int K, int N) {
  int idx = blockIdx.x * blockDim.x + threadIdx.x;
  if (idx >= K * N) return;
  int k = idx / N, n = idx % N;
  wT[(size_t)n * K + k] = (__bf16)w[(size_t)k * N + n];
}

// --------------------------- QKV projection --------------------------------
// Computes x @ w_qkv. Q,K rows -> qk[8192][2048]; V -> vT[4][1024][2048].
// One wave = 16x64 strip: 4 N-tiles sharing one A fragment per k-step.
__global__ void qkv_gemm_kernel(const float* __restrict__ x,
                                const __bf16* __restrict__ wT,
                                __bf16* __restrict__ qk,
                                __bf16* __restrict__ vT) {
  const int NG = N3C / 64;  // 48 n-groups
  int wid  = (blockIdx.x * blockDim.x + threadIdx.x) >> 5;
  int lane = threadIdx.x & 31;
  int hf = lane >> 4, ln = lane & 15;
  int ng = wid % NG, mt = wid / NG;

  const float* arow = x + (size_t)(mt * 16 + ln) * Cdim;
  const __bf16* bcol[4];
#pragma unroll
  for (int t = 0; t < 4; ++t)
    bcol[t] = wT + (size_t)(ng * 64 + t * 16 + ln) * Cdim;

  v8f acc[4] = {v8f{}, v8f{}, v8f{}, v8f{}};
#pragma unroll 2
  for (int k0 = 0; k0 < Cdim; k0 += 32) {
    v16bf a = load_fragA_f32(arow, k0, hf);
#pragma unroll
    for (int t = 0; t < 4; ++t)
      acc[t] = wmma_bf16(a, load_fragB_bf16(bcol[t], k0, hf), acc[t]);
  }

  int row0 = mt * 16 + 8 * hf;          // first of this lane's 8 rows
  int bidx = (mt * 16) >> 11;           // batch (rows never straddle)
  int l0   = row0 & (Lsq - 1);
#pragma unroll
  for (int t = 0; t < 4; ++t) {
    int col = ng * 64 + t * 16 + ln;
    if (col < C2) {                     // Q or K -> row-major qk
      __bf16* o = qk + (size_t)row0 * C2 + col;
#pragma unroll
      for (int r = 0; r < 8; ++r) o[(size_t)r * C2] = (__bf16)acc[t][r];
    } else {                            // V -> transposed vT[b][d][l]
      __bf16* o = vT + ((size_t)bidx * Cdim + (col - C2)) * Lsq + l0;
#pragma unroll
      for (int r = 0; r < 8; ++r) o[r] = (__bf16)acc[t][r];
    }
  }
}

// --------------------------- flash attention -------------------------------
// One wave per (b, h, 16-query tile). Online softmax over L=2048 keys,
// 32 keys per step. Q/K from qk (row-major), V from vT (key-contiguous).
__global__ void attn_kernel(const __bf16* __restrict__ qk,
                            const __bf16* __restrict__ vT,
                            __bf16* __restrict__ attn) {
  __shared__ __bf16 plds[8][16 * 32];   // per-wave P staging (C/D -> A relayout)
  int wslot = threadIdx.x >> 5;
  int wid   = (blockIdx.x * blockDim.x + threadIdx.x) >> 5;
  int lane  = threadIdx.x & 31;
  int hf = lane >> 4, ln = lane & 15;
  int qt = wid & 127, bh = wid >> 7, h = bh & 15, b = bh >> 4;
  int q0 = qt * 16;

  const __bf16* qbase = qk + (size_t)(b * Lsq + q0 + ln) * C2 + h * Dh;
  v16bf qa0 = load_fragA_bf16(qbase, 0, hf);
  v16bf qa1 = load_fragA_bf16(qbase, 32, hf);

  const __bf16* vcol[4];                // V^T rows: one d-column per lane
#pragma unroll
  for (int t = 0; t < 4; ++t)
    vcol[t] = vT + ((size_t)b * Cdim + h * Dh + t * 16 + ln) * Lsq;

  v8f acc[4] = {v8f{}, v8f{}, v8f{}, v8f{}};
  float mi[8], li[8];
#pragma unroll
  for (int r = 0; r < 8; ++r) { mi[r] = -1e30f; li[r] = 0.f; }

  __bf16* pp = &plds[wslot][0];

  for (int key0 = 0; key0 < Lsq; key0 += 32) {
    // --- S = Q @ K^T for 32 keys (two 16x16 tiles), contraction over D=64 ---
    const __bf16* kcol0 = qk + (size_t)(b * Lsq + key0 + ln) * C2 + Cdim + h * Dh;
    const __bf16* kcol1 = kcol0 + (size_t)16 * C2;
    v8f s0 = {}, s1 = {};
    s0 = wmma_bf16(qa0, load_fragB_bf16(kcol0, 0, hf), s0);
    s0 = wmma_bf16(qa1, load_fragB_bf16(kcol0, 32, hf), s0);
    s1 = wmma_bf16(qa0, load_fragB_bf16(kcol1, 0, hf), s1);
    s1 = wmma_bf16(qa1, load_fragB_bf16(kcol1, 32, hf), s1);

    // --- online softmax; row m = r + 8*hf, 16 columns across lanes 0..15 ---
#pragma unroll
    for (int r = 0; r < 8; ++r) {
      float tm = fmaxf(s0[r], s1[r]);
      tm = fmaxf(tm, __shfl_xor(tm, 1, 32));
      tm = fmaxf(tm, __shfl_xor(tm, 2, 32));
      tm = fmaxf(tm, __shfl_xor(tm, 4, 32));
      tm = fmaxf(tm, __shfl_xor(tm, 8, 32));
      float mn = fmaxf(mi[r], tm);
      float fz = __expf(mi[r] - mn);
      mi[r] = mn;
      li[r] *= fz;
      acc[0][r] *= fz; acc[1][r] *= fz; acc[2][r] *= fz; acc[3][r] *= fz;
      float p0 = __expf(s0[r] - mn);
      float p1 = __expf(s1[r] - mn);
      float ps = p0 + p1;
      ps += __shfl_xor(ps, 1, 32);
      ps += __shfl_xor(ps, 2, 32);
      ps += __shfl_xor(ps, 4, 32);
      ps += __shfl_xor(ps, 8, 32);
      li[r] += ps;
      int m = r + 8 * hf;
      pp[m * 32 + ln]      = (__bf16)p0;
      pp[m * 32 + 16 + ln] = (__bf16)p1;
    }
    asm volatile("s_wait_dscnt 0" ::: "memory");  // LDS writes -> A-frag reads

    v16bf pa = load_fragA_bf16(pp + ln * 32, 0, hf);

    // --- O += P @ V : 4 tiles over D=64; V^T rows are key-contiguous ---
#pragma unroll
    for (int t = 0; t < 4; ++t)
      acc[t] = wmma_bf16(pa, load_fragB_bf16(vcol[t], key0, hf), acc[t]);
  }

  // --- normalize and store bf16 attn output, row-major [8192][1024] ---
#pragma unroll
  for (int t = 0; t < 4; ++t) {
#pragma unroll
    for (int r = 0; r < 8; ++r) {
      int m = r + 8 * hf;
      attn[(size_t)(b * Lsq + q0 + m) * Cdim + h * Dh + t * 16 + ln] =
          (__bf16)(acc[t][r] / li[r]);
    }
  }
}

// --------------------------- output projection -----------------------------
// out[8192][1024] (f32) = attn (bf16) @ w_out + b_out ; wT is [1024][1024].
// One wave = 16x64 strip, shared A fragment.
__global__ void out_gemm_kernel(const __bf16* __restrict__ attn,
                                const __bf16* __restrict__ wT,
                                const float* __restrict__ bias,
                                float* __restrict__ out) {
  const int NG = Cdim / 64;  // 16 n-groups
  int wid  = (blockIdx.x * blockDim.x + threadIdx.x) >> 5;
  int lane = threadIdx.x & 31;
  int hf = lane >> 4, ln = lane & 15;
  int ng = wid % NG, mt = wid / NG;

  const __bf16* arow = attn + (size_t)(mt * 16 + ln) * Cdim;
  const __bf16* bcol[4];
#pragma unroll
  for (int t = 0; t < 4; ++t)
    bcol[t] = wT + (size_t)(ng * 64 + t * 16 + ln) * Cdim;

  v8f acc[4] = {v8f{}, v8f{}, v8f{}, v8f{}};
#pragma unroll 2
  for (int k0 = 0; k0 < Cdim; k0 += 32) {
    v16bf a = load_fragA_bf16(arow, k0, hf);
#pragma unroll
    for (int t = 0; t < 4; ++t)
      acc[t] = wmma_bf16(a, load_fragB_bf16(bcol[t], k0, hf), acc[t]);
  }

  int row0 = mt * 16 + 8 * hf;
#pragma unroll
  for (int t = 0; t < 4; ++t) {
    int col = ng * 64 + t * 16 + ln;
    float bv = bias[col];
    float* o = out + (size_t)row0 * Cdim + col;
#pragma unroll
    for (int r = 0; r < 8; ++r) o[(size_t)r * Cdim] = acc[t][r] + bv;
  }
}

// ---------------------------------------------------------------------------
extern "C" void kernel_launch(void* const* d_in, const int* in_sizes, int n_in,
                              void* d_out, int out_size, void* d_ws, size_t ws_size,
                              hipStream_t stream) {
  const float* x     = (const float*)d_in[0];  // [4,2048,1024]
  const float* w_qkv = (const float*)d_in[1];  // [1024,3072]
  const float* w_out = (const float*)d_in[2];  // [1024,1024]
  const float* b_out = (const float*)d_in[3];  // [1024]
  float* out = (float*)d_out;

  // Workspace (bytes): wqkvT(6M) | woutT(2M) | qk(32M) | vT(16M) | attn(16M)
  char* ws = (char*)d_ws;
  size_t off = 0;
  __bf16* wqkvT = (__bf16*)(ws + off); off += (size_t)N3C * Cdim * 2;
  __bf16* woutT = (__bf16*)(ws + off); off += (size_t)Cdim * Cdim * 2;
  __bf16* qk    = (__bf16*)(ws + off); off += (size_t)Mrows * C2 * 2;
  __bf16* vT    = (__bf16*)(ws + off); off += (size_t)Bsz * Cdim * Lsq * 2;
  __bf16* attn  = (__bf16*)(ws + off);

  // 1) weight convert + transpose to bf16 [N][K]
  convert_transpose_kernel<<<(Cdim * N3C + 255) / 256, 256, 0, stream>>>(
      w_qkv, wqkvT, Cdim, N3C);
  convert_transpose_kernel<<<(Cdim * Cdim + 255) / 256, 256, 0, stream>>>(
      w_out, woutT, Cdim, Cdim);

  // 2) QKV GEMM: 512 m-tiles x 48 n-groups, 8 waves/block
  {
    int waves = (Mrows / 16) * (N3C / 64);   // 24576
    qkv_gemm_kernel<<<waves / 8, 256, 0, stream>>>(x, wqkvT, qk, vT);
  }

  // 3) attention: 4*16*128 = 8192 wave-tasks
  attn_kernel<<<8192 / 8, 256, 0, stream>>>(qk, vT, attn);

  // 4) output GEMM + bias: 512 m-tiles x 16 n-groups
  {
    int waves = (Mrows / 16) * (Cdim / 64);  // 8192
    out_gemm_kernel<<<waves / 8, 256, 0, stream>>>(attn, woutT, b_out, out);
  }
}
